// TripletSTDP_68994354643156
// MI455X (gfx1250) — matrix operand order repaired and checked by reading, not compiled, not run on hardware
//
#include <hip/hip_runtime.h>

// ---------------------------------------------------------------------------
// TripletSTDP on MI455X (gfx1250):
//   Phase 1: per-(b,neuron) exponential trace scan over T, emitting f16
//            operand matrices into d_ws in k-block-tiled layout
//            [KDIM/32][1024 rows][32]  (64 B per row per k-block).
//   Phase 2: zero d_out.
//   Phase 3: fused dual-GEMM  WU = U * Post^T + Pre * (-V)^T via
//            v_wmma_f32_16x16x32_f16; K split across gridDim.z; f32
//            atomicAdd into d_out scaled by 1/(B*T).
//
// k-block-tiled operand layout => every WMMA fragment load is part of a
// contiguous 1 KB-per-wave region (full cache-line utilization).
// __launch_bounds__(128, 1) gives the allocator the full VGPR budget so all
// 16 fragments stay live in distinct registers (pipelined loads, no hazard
// NOPs from operand-range recycling).
// ---------------------------------------------------------------------------

typedef __attribute__((ext_vector_type(16))) _Float16 v16h;
typedef __attribute__((ext_vector_type(8)))  _Float16 v8h;
typedef __attribute__((ext_vector_type(8)))  float    v8f;

#define BATCH   32
#define TSTEPS  512
#define NPRE    1024
#define NPOST   1024
#define KDIM    (BATCH * TSTEPS)      // 16384
#define KSLICES 8
#define KCHUNK  (KDIM / KSLICES)      // 2048
#define NROWS   1024                  // rows per operand matrix
#define KBLK_STRIDE ((size_t)NROWS * 32)   // elements per k-block (32768)
#define PF_DIST 8                     // prefetch distance in k-blocks

// exp(-1/tau) decay constants
#define D_PLUS  0.95122942450071400910f   // exp(-1/20)
#define D_X     0.99014786265815465900f   // exp(-1/101)
#define D_MINUS 0.95122942450071400910f   // exp(-1/20)
#define D_Y     0.99126643123569500000f   // exp(-1/114)

#define A_PLUS   0.005f
#define A_MINUS  0.00525f
#define A_TPLUS  0.0001f
#define A_TMINUS 0.0001f

// ---------------------------------------------------------------------------
// Phase 1a: pre-synaptic traces.
// Block = (32,32): tx = neuron-in-tile (read lane), ty = batch index.
// Reads fp32 pre[b, t, p] coalesced (lane = p); after an LDS 32x32 transpose,
// writes f16 into k-block-tiled layout: element (row p, k=t*32+b) lives at
//   t*KBLK_STRIDE + p*32 + b
// so each t-step's block write is one contiguous 2 KB chunk per matrix.
// ---------------------------------------------------------------------------
__global__ __launch_bounds__(1024) void stdp_pre_traces(
    const float* __restrict__ pre,
    _Float16* __restrict__ uT,
    _Float16* __restrict__ preT)
{
    const int tx = threadIdx.x;           // 0..31
    const int ty = threadIdx.y;           // 0..31
    const int p_read = blockIdx.x * 32 + tx;
    const int b      = ty;

    __shared__ _Float16 lu[32][33];
    __shared__ _Float16 lp[32][33];

    float r1 = 0.f, r2 = 0.f;
    const size_t in_base   = (size_t)b * TSTEPS * NPRE + p_read;
    const size_t row_write = (size_t)(blockIdx.x * 32 + ty) * 32 + tx;

    for (int t = 0; t < TSTEPS; ++t) {
        r1 *= D_PLUS;
        r2 *= D_X;
        const float u = A_PLUS * r1 + A_TPLUS * (r1 * r2);
        const float s = pre[in_base + (size_t)t * NPRE];
        lu[tx][b] = (_Float16)u;
        lp[tx][b] = (_Float16)s;
        r1 += s;
        r2 += s;
        __syncthreads();
        const size_t o = (size_t)t * KBLK_STRIDE + row_write;
        uT  [o] = lu[ty][tx];
        preT[o] = lp[ty][tx];
        __syncthreads();
    }
}

// ---------------------------------------------------------------------------
// Phase 1b: post-synaptic traces (V stored NEGATED so the GEMM only adds).
// ---------------------------------------------------------------------------
__global__ __launch_bounds__(1024) void stdp_post_traces(
    const float* __restrict__ post,
    _Float16* __restrict__ vT,
    _Float16* __restrict__ postT)
{
    const int tx = threadIdx.x;
    const int ty = threadIdx.y;
    const int q_read = blockIdx.x * 32 + tx;
    const int b      = ty;

    __shared__ _Float16 lv[32][33];
    __shared__ _Float16 ls[32][33];

    float o1 = 0.f, o2 = 0.f;
    const size_t in_base   = (size_t)b * TSTEPS * NPOST + q_read;
    const size_t row_write = (size_t)(blockIdx.x * 32 + ty) * 32 + tx;

    for (int t = 0; t < TSTEPS; ++t) {
        o1 *= D_MINUS;
        o2 *= D_Y;
        const float v = -(A_MINUS * o1 + A_TMINUS * (o1 * o2));
        const float s = post[in_base + (size_t)t * NPOST];
        lv[tx][b] = (_Float16)v;
        ls[tx][b] = (_Float16)s;
        o1 += s;
        o2 += s;
        __syncthreads();
        const size_t o = (size_t)t * KBLK_STRIDE + row_write;
        vT   [o] = lv[ty][tx];
        postT[o] = ls[ty][tx];
        __syncthreads();
    }
}

// ---------------------------------------------------------------------------
// Phase 2: zero the output accumulator.
// ---------------------------------------------------------------------------
__global__ void stdp_zero(float* __restrict__ out, int n)
{
    int i = blockIdx.x * blockDim.x + threadIdx.x;
    if (i < n) out[i] = 0.f;
}

// ---------------------------------------------------------------------------
// Phase 3: fused dual WMMA GEMM.
// Block: 128 threads = 4 waves (2x2), wave tile 64x64 -> block tile 128x128.
// grid = (NPOST/128, NPRE/128, KSLICES).
//
// Operand storage: [k_block][row][32] f16, row chunk = 64 B.
// Fragment addressing (CDNA5 wave32 WMMA 16x16x32 f16 layouts):
//   A (16x32): lane = M | (half<<4); chunks at K = half*8 and 16+half*8
//       -> bytes row*64 + half*16 (16 B) and row*64 + 32 + half*16 (16 B)
//   B (32x16): lane = N | (half<<4); chunks at K = half*8 and 16+half*8
//       -> bytes row*64 + half*16 (16 B) and row*64 + 32 + half*16 (16 B)
//   C/D: 8 VGPRs; VGPR r holds M = half*8 + r, N = lane&15.
// ---------------------------------------------------------------------------
__global__ __launch_bounds__(128, 1) void stdp_wmma_gemm(
    const _Float16* __restrict__ uT,    // tiled [KDIM/32][NPRE ][32]
    const _Float16* __restrict__ postT, // tiled [KDIM/32][NPOST][32]
    const _Float16* __restrict__ preT,  // tiled [KDIM/32][NPRE ][32]
    const _Float16* __restrict__ vT,    // tiled [KDIM/32][NPOST][32] (negated)
    float* __restrict__ out)            // [NPRE][NPOST]
{
    const int tid  = threadIdx.x;
    const int lane = tid & 31;
    const int wave = tid >> 5;                  // 0..3
    const int wrow = wave >> 1;                 // 0..1
    const int wcol = wave & 1;                  // 0..1

    const int m_blk  = blockIdx.y * 128;
    const int n_blk  = blockIdx.x * 128;
    const int m_base = m_blk + wrow * 64;
    const int n_base = n_blk + wcol * 64;
    const int kb_begin = blockIdx.z * (KCHUNK / 32);   // k-block index
    const int kb_end   = kb_begin + (KCHUNK / 32);

    const int half = lane >> 4;                 // 0 or 1
    const int l16  = lane & 15;
    const int a_koff = half * 8;                // K offset of A chunk 0
    const int b_koff = half * 8;                // K offset of B chunk 0

    v8f acc[4][4];
#pragma unroll
    for (int mi = 0; mi < 4; ++mi)
#pragma unroll
        for (int ni = 0; ni < 4; ++ni)
            acc[mi][ni] = (v8f){0.f, 0.f, 0.f, 0.f, 0.f, 0.f, 0.f, 0.f};

    for (int kb = kb_begin; kb < kb_end; ++kb) {
        const size_t kbase = (size_t)kb * KBLK_STRIDE;

        // Prefetch the k-block PF_DIST ahead: one 64 B row per thread covers
        // the full 8 KB/matrix tile for this block (global_prefetch path).
        if (kb + PF_DIST < kb_end) {
            const size_t pbase = (size_t)(kb + PF_DIST) * KBLK_STRIDE;
            const size_t aofs  = pbase + ((size_t)m_blk + tid) * 32;
            const size_t bofs  = pbase + ((size_t)n_blk + tid) * 32;
            __builtin_prefetch(uT + aofs, 0, 1);
            __builtin_prefetch(preT + aofs, 0, 1);
            __builtin_prefetch(postT + bofs, 0, 1);
            __builtin_prefetch(vT + bofs, 0, 1);
        }

        v16h a1[4], a2[4], b1[4], b2[4];

#pragma unroll
        for (int mi = 0; mi < 4; ++mi) {
            const size_t roff =
                kbase + (size_t)(m_base + mi * 16 + l16) * 32 + a_koff;
            const v8h lo1 = *(const v8h*)(uT + roff);
            const v8h hi1 = *(const v8h*)(uT + roff + 16);
            const v8h lo2 = *(const v8h*)(preT + roff);
            const v8h hi2 = *(const v8h*)(preT + roff + 16);
#pragma unroll
            for (int i = 0; i < 8; ++i) {
                a1[mi][i]     = lo1[i];
                a1[mi][i + 8] = hi1[i];
                a2[mi][i]     = lo2[i];
                a2[mi][i + 8] = hi2[i];
            }
        }

#pragma unroll
        for (int ni = 0; ni < 4; ++ni) {
            const size_t roff =
                kbase + (size_t)(n_base + ni * 16 + l16) * 32 + b_koff;
            const v8h lo1 = *(const v8h*)(postT + roff);
            const v8h hi1 = *(const v8h*)(postT + roff + 16);
            const v8h lo2 = *(const v8h*)(vT + roff);
            const v8h hi2 = *(const v8h*)(vT + roff + 16);
#pragma unroll
            for (int i = 0; i < 8; ++i) {
                b1[ni][i]     = lo1[i];
                b1[ni][i + 8] = hi1[i];
                b2[ni][i]     = lo2[i];
                b2[ni][i + 8] = hi2[i];
            }
        }

#pragma unroll
        for (int mi = 0; mi < 4; ++mi) {
#pragma unroll
            for (int ni = 0; ni < 4; ++ni) {
                acc[mi][ni] = __builtin_amdgcn_wmma_f32_16x16x32_f16(
                    false, a1[mi], false, b1[ni],
                    (short)0, acc[mi][ni], false, false);
                acc[mi][ni] = __builtin_amdgcn_wmma_f32_16x16x32_f16(
                    false, a2[mi], false, b2[ni],
                    (short)0, acc[mi][ni], false, false);
            }
        }
    }

    const float scale = 1.0f / (float)(BATCH * TSTEPS);   // 1/16384
#pragma unroll
    for (int mi = 0; mi < 4; ++mi) {
#pragma unroll
        for (int ni = 0; ni < 4; ++ni) {
            const int m0 = m_base + mi * 16 + half * 8;
            const int n  = n_base + ni * 16 + l16;
#pragma unroll
            for (int r = 0; r < 8; ++r) {
                atomicAdd(&out[(size_t)(m0 + r) * NPOST + n],
                          acc[mi][ni][r] * scale);
            }
        }
    }
}

// ---------------------------------------------------------------------------
// Host launch. d_in: [0]=pre_spikes f32 [B,T,NPRE], [1]=post_spikes f32
// [B,T,NPOST], [2]=weights f32 (shape-only, unused). d_out: f32 [NPRE,NPOST].
// Workspace: 4 f16 matrices of 1024*16384 elements = 128 MB total.
// ---------------------------------------------------------------------------
extern "C" void kernel_launch(void* const* d_in, const int* in_sizes, int n_in,
                              void* d_out, int out_size, void* d_ws, size_t ws_size,
                              hipStream_t stream)
{
    (void)in_sizes; (void)n_in; (void)out_size; (void)ws_size;

    const float* pre  = (const float*)d_in[0];
    const float* post = (const float*)d_in[1];
    float* out = (float*)d_out;

    const size_t mat = (size_t)NROWS * KDIM;   // elements per f16 matrix
    _Float16* ws    = (_Float16*)d_ws;
    _Float16* uT    = ws;
    _Float16* preT  = ws + mat;
    _Float16* postT = ws + 2 * mat;
    _Float16* vT    = ws + 3 * mat;

    dim3 tb(32, 32);
    stdp_pre_traces <<<dim3(NPRE  / 32), tb, 0, stream>>>(pre,  uT, preT);
    stdp_post_traces<<<dim3(NPOST / 32), tb, 0, stream>>>(post, vT, postT);

    const int nOut = NPRE * NPOST;
    stdp_zero<<<(nOut + 255) / 256, 256, 0, stream>>>(out, nOut);

    dim3 grid(NPOST / 128, NPRE / 128, KSLICES);
    stdp_wmma_gemm<<<grid, 128, 0, stream>>>(uT, postT, preT, vT, out);
}
